// TrackFormerDecoderLayer_86114094285177
// MI455X (gfx1250) — matrix operand order repaired and checked by reading, not compiled, not run on hardware
//
#include <hip/hip_runtime.h>

// ---------------------------------------------------------------------------
// Types / helpers
// ---------------------------------------------------------------------------
typedef __attribute__((ext_vector_type(16))) __bf16 v16bf;
typedef __attribute__((ext_vector_type(2)))  __bf16 v2bf;
typedef __attribute__((ext_vector_type(8)))  float  v8f;
typedef __attribute__((ext_vector_type(4)))  float  v4f;

#define NBATCH 8
#define QN     900
#define DIM    256
#define NHEAD  8
#define HDIM   32
#define PPTS   4
#define DFF    512
#define HBV    200
#define WBV    200
#define ROWS   (NBATCH * QN)      // 7200
#define HW     (HBV * WBV)        // 40000

__device__ __forceinline__ int imin(int a, int b) { return a < b ? a : b; }
__device__ __forceinline__ int imax(int a, int b) { return a > b ? a : b; }

// native f32 -> bf16 conversion (hardware cvt on gfx1250)
__device__ __forceinline__ __bf16 f2bf(float f) { return (__bf16)f; }
__device__ __forceinline__ unsigned short f2bfu(float f) {
  return __builtin_bit_cast(unsigned short, (__bf16)f);
}
// pack two floats into a bf16x2 dword (k-adjacent pair)
__device__ __forceinline__ unsigned pack2bf(float a, float b) {
  v2bf v;
  v[0] = (__bf16)a;
  v[1] = (__bf16)b;
  return __builtin_bit_cast(unsigned, v);
}
__device__ __forceinline__ float bfu2f(unsigned short s) {
  return __builtin_bit_cast(float, (unsigned)s << 16);
}

// ---------------------------------------------------------------------------
// Wide WMMA GEMM: C[M,N] = act(A[M,K] @ W[K,N] + bias)
// block = 256 threads (8 waves); tile BM=32 x BN=128; wave computes 16x32
// (two v_wmma sharing one A fragment). grid = (M/32, N/128).
// Requires M%32==0, N%128==0, K%32==0.
// NT_A: non-temporal loads of A (streaming input; keep L2 for resident data).
// ---------------------------------------------------------------------------
template <bool RELU, bool OUT_BF16, bool NT_A>
__global__ __launch_bounds__(256) void gemm_wide_kernel(
    const float* __restrict__ A, const float* __restrict__ W,
    const float* __restrict__ bias, void* __restrict__ Cout,
    int M, int N, int K) {
  __shared__ __bf16 As[32][32];      // [m][k]
  __shared__ __bf16 Bs[128][32];     // transposed: [n][k]

  const int t    = threadIdx.x;
  const int lane = t & 31;
  const int wave = t >> 5;                 // 0..7
  const int wm   = (wave >> 2) * 16;       // 0 or 16
  const int wn   = (wave & 3) * 32;        // 0,32,64,96
  const int m0   = blockIdx.x * 32;
  const int n0   = blockIdx.y * 128;

  // A staging: 1024 elems / 256 thr = 4 each (one row, 4 cols)
  const int ar = t >> 3;            // 0..31
  const int ac = (t & 7) * 4;       // 0..28
  // B staging: 4096 elems / 256 thr = 16 each (2 k-rows x 8 n-cols)
  const int brp = t >> 4;           // k-row pair 0..15  -> rows 2*brp, 2*brp+1
  const int bcg = (t & 15) * 8;     // n-col group 0..120

  v8f acc0 = {}, acc1 = {};

  for (int k0 = 0; k0 < K; k0 += 32) {
    // stage A (fp32 -> bf16, packed pair stores)
    {
      const float* src = A + (size_t)(m0 + ar) * K + (k0 + ac);
      v4f f;
      if (NT_A)
        f = __builtin_nontemporal_load((const v4f*)src);
      else
        f = *(const v4f*)src;
      unsigned* ad = (unsigned*)&As[ar][ac];
      ad[0] = pack2bf(f[0], f[1]);
      ad[1] = pack2bf(f[2], f[3]);
      const float* nxt = A + (size_t)(m0 + ar) * K + (imin(k0 + 32, K - 32) + ac);
      __builtin_prefetch(nxt, 0, 3);
    }
    // stage B transposed (two k-rows per thread -> packed k-pair dword stores)
    {
      const float* s0 = W + (size_t)(k0 + 2 * brp) * N + (n0 + bcg);
      const float* s1 = s0 + N;
      v4f r00 = *(const v4f*)(s0);
      v4f r01 = *(const v4f*)(s0 + 4);
      v4f r10 = *(const v4f*)(s1);
      v4f r11 = *(const v4f*)(s1 + 4);
      unsigned* bd0 = (unsigned*)&Bs[bcg + 0][2 * brp];
      // row stride in dwords between n columns: 32 bf16 = 16 dwords
      bd0[0 * 16] = pack2bf(r00[0], r10[0]);
      bd0[1 * 16] = pack2bf(r00[1], r10[1]);
      bd0[2 * 16] = pack2bf(r00[2], r10[2]);
      bd0[3 * 16] = pack2bf(r00[3], r10[3]);
      bd0[4 * 16] = pack2bf(r01[0], r11[0]);
      bd0[5 * 16] = pack2bf(r01[1], r11[1]);
      bd0[6 * 16] = pack2bf(r01[2], r11[2]);
      bd0[7 * 16] = pack2bf(r01[3], r11[3]);
      const float* nxt = W + (size_t)(imin(k0 + 32, K - 32) + 2 * brp) * N + (n0 + bcg);
      __builtin_prefetch(nxt, 0, 3);
    }
    __syncthreads();

    // A fragment: lane-half selects K base {0,16}; K in kb..kb+15
    const int ml = wm + (lane & 15);
    const int kb = (lane >> 4) * 16;
    v16bf af;
    {
      const __bf16* p0 = &As[ml][kb];
      const __bf16* p1 = &As[ml][kb + 8];
#pragma unroll
      for (int i = 0; i < 8; ++i) { af[i] = p0[i]; af[i + 8] = p1[i]; }
    }
    // B fragments: 16 consecutive K for fixed N column
    const int kb2 = (lane >> 4) * 16;
    v16bf bf0, bf1;
    {
      const __bf16* p0 = &Bs[wn + (lane & 15)][kb2];
      const __bf16* p1 = &Bs[wn + 16 + (lane & 15)][kb2];
#pragma unroll
      for (int i = 0; i < 16; ++i) { bf0[i] = p0[i]; bf1[i] = p1[i]; }
    }
    acc0 = __builtin_amdgcn_wmma_f32_16x16x32_bf16(false, af, false, bf0,
                                                   (short)0, acc0, false, false);
    acc1 = __builtin_amdgcn_wmma_f32_16x16x32_bf16(false, af, false, bf1,
                                                   (short)0, acc1, false, false);
    __syncthreads();
  }

  // epilogue: D layout VGPR r -> M = r + 8*(lane>=16), N = lane&15
  const int col0 = n0 + wn + (lane & 15);
  const int col1 = col0 + 16;
  const float bv0 = bias ? bias[col0] : 0.0f;
  const float bv1 = bias ? bias[col1] : 0.0f;
#pragma unroll
  for (int r = 0; r < 8; ++r) {
    int row = m0 + wm + (lane >> 4) * 8 + r;
    float v0 = acc0[r] + bv0;
    float v1 = acc1[r] + bv1;
    if (RELU) { v0 = fmaxf(v0, 0.0f); v1 = fmaxf(v1, 0.0f); }
    if (OUT_BF16) {
      ((unsigned short*)Cout)[(size_t)row * N + col0] = f2bfu(v0);
      ((unsigned short*)Cout)[(size_t)row * N + col1] = f2bfu(v1);
    } else {
      ((float*)Cout)[(size_t)row * N + col0] = v0;
      ((float*)Cout)[(size_t)row * N + col1] = v1;
    }
  }
}

// ---------------------------------------------------------------------------
// Small WMMA GEMM (for N=64 / N=32 projections): block 128 (4 waves),
// tile 32x32, one 16x16 tile per wave. grid = (M/32, N/32).
// ---------------------------------------------------------------------------
__global__ __launch_bounds__(128) void gemm_small_kernel(
    const float* __restrict__ A, const float* __restrict__ W,
    const float* __restrict__ bias, float* __restrict__ Cout,
    int M, int N, int K) {
  __shared__ __bf16 As[32][32];
  __shared__ __bf16 Bs[32][32];   // [n][k]

  const int t    = threadIdx.x;
  const int lane = t & 31;
  const int wave = t >> 5;
  const int wm   = (wave >> 1) * 16;
  const int wn   = (wave & 1) * 16;
  const int m0   = blockIdx.x * 32;
  const int n0   = blockIdx.y * 32;

  // A staging: 1024/128 = 8 each (one row, 8 cols)
  const int ar = t >> 2;            // 0..31
  const int ac = (t & 3) * 8;       // 0,8,16,24
  // B staging: 1024/128 = 8 each (2 k-rows x 4 n-cols)
  const int brp = t >> 3;           // 0..15
  const int bcg = (t & 7) * 4;      // 0..28

  v8f acc = {};
  for (int k0 = 0; k0 < K; k0 += 32) {
    {
      const float* src = A + (size_t)(m0 + ar) * K + (k0 + ac);
      v4f f0 = *(const v4f*)(src);
      v4f f1 = *(const v4f*)(src + 4);
      unsigned* ad = (unsigned*)&As[ar][ac];
      ad[0] = pack2bf(f0[0], f0[1]);
      ad[1] = pack2bf(f0[2], f0[3]);
      ad[2] = pack2bf(f1[0], f1[1]);
      ad[3] = pack2bf(f1[2], f1[3]);
    }
    {
      const float* s0 = W + (size_t)(k0 + 2 * brp) * N + (n0 + bcg);
      const float* s1 = s0 + N;
      v4f r0 = *(const v4f*)(s0);
      v4f r1 = *(const v4f*)(s1);
      unsigned* bd = (unsigned*)&Bs[bcg][2 * brp];
      // row stride between n columns: 32 bf16 = 16 dwords
      bd[0 * 16] = pack2bf(r0[0], r1[0]);
      bd[1 * 16] = pack2bf(r0[1], r1[1]);
      bd[2 * 16] = pack2bf(r0[2], r1[2]);
      bd[3 * 16] = pack2bf(r0[3], r1[3]);
    }
    __syncthreads();
    const int ml = wm + (lane & 15);
    const int kb = (lane >> 4) * 16;
    v16bf af;
    {
      const __bf16* p0 = &As[ml][kb];
      const __bf16* p1 = &As[ml][kb + 8];
#pragma unroll
      for (int i = 0; i < 8; ++i) { af[i] = p0[i]; af[i + 8] = p1[i]; }
    }
    const int kb2 = (lane >> 4) * 16;
    v16bf bf;
    {
      const __bf16* p = &Bs[wn + (lane & 15)][kb2];
#pragma unroll
      for (int i = 0; i < 16; ++i) bf[i] = p[i];
    }
    acc = __builtin_amdgcn_wmma_f32_16x16x32_bf16(false, af, false, bf,
                                                  (short)0, acc, false, false);
    __syncthreads();
  }
  const int col = n0 + wn + (lane & 15);
  const float bv = bias ? bias[col] : 0.0f;
#pragma unroll
  for (int r = 0; r < 8; ++r) {
    int row = m0 + wm + (lane >> 4) * 8 + r;
    Cout[(size_t)row * N + col] = acc[r] + bv;
  }
}

// ---------------------------------------------------------------------------
// Self-attention: one wave handles 16 query rows for one (batch, head).
// scores (16 x 928 padded) kept in LDS; WMMA for QK^T and P*V.
// All guarded loads are branchless: clamped address + 0/1 mask multiply.
// grid = (57, 64), block = 32.
// ---------------------------------------------------------------------------
#define KTILES 57          // ceil(900/16)
#define SW     928         // 29*32 padded score width

__global__ __launch_bounds__(32) void attn_kernel(
    const float* __restrict__ qp, const float* __restrict__ kp,
    const float* __restrict__ vp, float* __restrict__ out) {
  __shared__ float sc[16][SW];          // 59392 B

  const int bh    = blockIdx.y;
  const int b     = bh >> 3;
  const int h     = bh & 7;
  const int qc    = blockIdx.x;
  const int lane  = threadIdx.x;
  const int qbase = qc * 16;
  const float scale = 0.17677669529663689f;   // 1/sqrt(32)

  const int ml = lane & 15;
  const int kb = (lane >> 4) * 16;

  // Q fragment (A, 16x32), scale folded in; reused across key tiles
  v16bf aq;
  {
    const int qrow = qbase + ml;
    const float qs = (qrow < QN) ? scale : 0.0f;
    const float* qptr =
        qp + (size_t)(b * QN + imin(qrow, QN - 1)) * DIM + h * HDIM;
#pragma unroll
    for (int i = 0; i < 8; ++i) {
      aq[i]     = f2bf(qptr[kb + i] * qs);
      aq[i + 8] = f2bf(qptr[kb + 8 + i] * qs);
    }
  }

  // scores = Q @ K^T
  for (int kt = 0; kt < KTILES; ++kt) {
    const int kcol = kt * 16 + (lane & 15);
    const float kz = (kcol < QN) ? 1.0f : 0.0f;
    const float* kptr = kp + (size_t)(b * QN + imin(kcol, QN - 1)) * DIM +
                        h * HDIM + (lane >> 4) * 16;
    v16bf bk;
#pragma unroll
    for (int i = 0; i < 16; ++i) bk[i] = f2bf(kptr[i] * kz);

    v8f c = {};
    c = __builtin_amdgcn_wmma_f32_16x16x32_bf16(false, aq, false, bk,
                                                (short)0, c, false, false);
#pragma unroll
    for (int r = 0; r < 8; ++r) {
      int row = r + (lane >> 4) * 8;
      int cix = kt * 16 + (lane & 15);
      sc[row][cix] = (cix < QN) ? c[r] : -1e30f;
    }
  }
  // pad columns 912..927
  for (int cix = 912 + lane; cix < SW; cix += 32)
#pragma unroll
    for (int r = 0; r < 16; ++r) sc[r][cix] = -1e30f;
  __syncthreads();

  // row softmax over padded width
  for (int r = 0; r < 16; ++r) {
    float mx = -1e30f;
    for (int c = lane; c < SW; c += 32) mx = fmaxf(mx, sc[r][c]);
#pragma unroll
    for (int o = 16; o; o >>= 1) mx = fmaxf(mx, __shfl_xor(mx, o, 32));
    float sm = 0.0f;
    for (int c = lane; c < SW; c += 32) {
      float e = __expf(sc[r][c] - mx);
      sc[r][c] = e;
      sm += e;
    }
#pragma unroll
    for (int o = 16; o; o >>= 1) sm += __shfl_xor(sm, o, 32);
    float inv = 1.0f / sm;
    for (int c = lane; c < SW; c += 32) sc[r][c] *= inv;
  }
  __syncthreads();

  // out = P @ V  (two 16-wide N tiles cover HDIM=32)
  v8f c0 = {}, c1 = {};
  for (int kc = 0; kc < SW / 32; ++kc) {
    v16bf ap;
    {
      const float* p0 = &sc[lane & 15][kc * 32 + kb];
      const float* p1 = &sc[lane & 15][kc * 32 + kb + 8];
#pragma unroll
      for (int i = 0; i < 8; ++i) { ap[i] = f2bf(p0[i]); ap[i + 8] = f2bf(p1[i]); }
    }
    const int krow0 = kc * 32 + (lane >> 4) * 16;
    const int ncol  = lane & 15;
    v16bf b0, b1;
#pragma unroll
    for (int i = 0; i < 16; ++i) {
      int vr = krow0 + i;
      float z = (vr < QN) ? 1.0f : 0.0f;
      const float* vpr =
          vp + (size_t)(b * QN + imin(vr, QN - 1)) * DIM + h * HDIM;
      b0[i] = f2bf(vpr[ncol] * z);
      b1[i] = f2bf(vpr[16 + ncol] * z);
    }
    c0 = __builtin_amdgcn_wmma_f32_16x16x32_bf16(false, ap, false, b0,
                                                 (short)0, c0, false, false);
    c1 = __builtin_amdgcn_wmma_f32_16x16x32_bf16(false, ap, false, b1,
                                                 (short)0, c1, false, false);
  }
#pragma unroll
  for (int r = 0; r < 8; ++r) {
    int qrow2 = qbase + (lane >> 4) * 8 + r;
    if (qrow2 < QN) {
      float* op = out + (size_t)(b * QN + qrow2) * DIM + h * HDIM;
      op[lane & 15]        = c0[r];
      op[16 + (lane & 15)] = c1[r];
    }
  }
}

// ---------------------------------------------------------------------------
// Deformable sampling: one wave per (b,q,h); lane = head-dim channel.
// Inline softmax over P=4; 4-corner bilinear gather from bf16 value (N,HW,D).
// Branchless boundary handling: clamp index, multiply by validity.
// ---------------------------------------------------------------------------
__global__ __launch_bounds__(128) void deform_sample_kernel(
    const float* __restrict__ ref, const float* __restrict__ off,
    const float* __restrict__ awl, const unsigned short* __restrict__ value,
    float* __restrict__ out) {
  const int wid  = blockIdx.x * 4 + (threadIdx.x >> 5);
  const int lane = threadIdx.x & 31;
  if (wid >= ROWS * NHEAD) return;
  const int h   = wid & 7;
  const int row = wid >> 3;        // b*QN + q
  const int b   = row / QN;

  const float rx = ref[row * 2 + 0];
  const float ry = ref[row * 2 + 1];
  const float* op = off + (size_t)row * (NHEAD * PPTS * 2) + h * (PPTS * 2);
  const float* ap = awl + (size_t)row * (NHEAD * PPTS) + h * PPTS;

  float l0 = ap[0], l1 = ap[1], l2 = ap[2], l3 = ap[3];
  float mx = fmaxf(fmaxf(l0, l1), fmaxf(l2, l3));
  float e0 = __expf(l0 - mx), e1 = __expf(l1 - mx);
  float e2 = __expf(l2 - mx), e3 = __expf(l3 - mx);
  float inv = 1.0f / (e0 + e1 + e2 + e3);
  float w[PPTS] = {e0 * inv, e1 * inv, e2 * inv, e3 * inv};

  const unsigned short* vbase =
      value + ((size_t)b * HW) * DIM + h * HDIM + lane;

  float acc = 0.0f;
#pragma unroll
  for (int p = 0; p < PPTS; ++p) {
    float px = (rx + op[p * 2 + 0] * (1.0f / WBV)) * WBV - 0.5f;
    float py = (ry + op[p * 2 + 1] * (1.0f / HBV)) * HBV - 0.5f;
    float x0f = floorf(px), y0f = floorf(py);
    float wx = px - x0f, wy = py - y0f;
    int x0 = (int)x0f, y0 = (int)y0f;
    float sp = 0.0f;
#pragma unroll
    for (int dy = 0; dy < 2; ++dy)
#pragma unroll
      for (int dx = 0; dx < 2; ++dx) {
        int xi = x0 + dx, yi = y0 + dy;
        int xc = imin(imax(xi, 0), WBV - 1);
        int yc = imin(imax(yi, 0), HBV - 1);
        float vld = (xi == xc && yi == yc) ? 1.0f : 0.0f;   // in-bounds
        float wgt = (dx ? wx : 1.0f - wx) * (dy ? wy : 1.0f - wy);
        sp += wgt * vld * bfu2f(vbase[(size_t)(yc * WBV + xc) * DIM]);
      }
    acc += w[p] * sp;
  }
  out[(size_t)row * DIM + h * HDIM + lane] = acc;
}

// ---------------------------------------------------------------------------
// Fused residual + LayerNorm over 256-wide rows; one wave per row.
// ---------------------------------------------------------------------------
__global__ __launch_bounds__(256) void resid_ln_kernel(
    const float* __restrict__ x, const float* __restrict__ r,
    const float* __restrict__ g, const float* __restrict__ bta,
    float* __restrict__ y, int rows) {
  const int wave = threadIdx.x >> 5;
  const int lane = threadIdx.x & 31;
  const int row  = blockIdx.x * 8 + wave;
  if (row >= rows) return;
  const float* xp = x + (size_t)row * DIM;
  const float* rp = r + (size_t)row * DIM;

  float v[8];
  float s = 0.0f;
#pragma unroll
  for (int i = 0; i < 8; ++i) {
    int c = lane + i * 32;
    v[i] = xp[c] + rp[c];
    s += v[i];
  }
#pragma unroll
  for (int o = 16; o; o >>= 1) s += __shfl_xor(s, o, 32);
  float mean = s * (1.0f / DIM);
  float q = 0.0f;
#pragma unroll
  for (int i = 0; i < 8; ++i) {
    float d = v[i] - mean;
    q += d * d;
  }
#pragma unroll
  for (int o = 16; o; o >>= 1) q += __shfl_xor(q, o, 32);
  float rstd = rsqrtf(q * (1.0f / DIM) + 1e-5f);
#pragma unroll
  for (int i = 0; i < 8; ++i) {
    int c = lane + i * 32;
    y[(size_t)row * DIM + c] = (v[i] - mean) * rstd * g[c] + bta[c];
  }
}

// ---------------------------------------------------------------------------
// Orchestration
// ---------------------------------------------------------------------------
extern "C" void kernel_launch(void* const* d_in, const int* in_sizes, int n_in,
                              void* d_out, int out_size, void* d_ws, size_t ws_size,
                              hipStream_t stream) {
  (void)in_sizes; (void)n_in; (void)out_size; (void)ws_size;
  const float* queries  = (const float*)d_in[0];
  const float* bev      = (const float*)d_in[1];
  const float* ref      = (const float*)d_in[2];
  const float* sa_wq    = (const float*)d_in[3];
  const float* sa_bq    = (const float*)d_in[4];
  const float* sa_wk    = (const float*)d_in[5];
  const float* sa_bk    = (const float*)d_in[6];
  const float* sa_wv    = (const float*)d_in[7];
  const float* sa_bv    = (const float*)d_in[8];
  const float* sa_wo    = (const float*)d_in[9];
  const float* sa_bo    = (const float*)d_in[10];
  const float* ln1_g    = (const float*)d_in[11];
  const float* ln1_b    = (const float*)d_in[12];
  const float* da_vw    = (const float*)d_in[13];
  const float* da_vb    = (const float*)d_in[14];
  const float* da_ow    = (const float*)d_in[15];
  const float* da_ob    = (const float*)d_in[16];
  const float* da_aw    = (const float*)d_in[17];
  const float* da_ab    = (const float*)d_in[18];
  const float* da_out_w = (const float*)d_in[19];
  const float* da_out_b = (const float*)d_in[20];
  const float* ln2_g    = (const float*)d_in[21];
  const float* ln2_b    = (const float*)d_in[22];
  const float* ff_w1    = (const float*)d_in[23];
  const float* ff_b1    = (const float*)d_in[24];
  const float* ff_w2    = (const float*)d_in[25];
  const float* ff_b2    = (const float*)d_in[26];
  const float* ln3_g    = (const float*)d_in[27];
  const float* ln3_b    = (const float*)d_in[28];

  char* wsp = (char*)d_ws;
  auto alloc = [&](size_t bytes) -> void* {
    void* p = wsp;
    wsp += (bytes + 255) & ~(size_t)255;
    return p;
  };
  float* qp       = (float*)alloc((size_t)ROWS * DIM * 4);
  float* kp       = (float*)alloc((size_t)ROWS * DIM * 4);
  float* vp       = (float*)alloc((size_t)ROWS * DIM * 4);
  float* attn_out = (float*)alloc((size_t)ROWS * DIM * 4);
  float* sa_proj  = (float*)alloc((size_t)ROWS * DIM * 4);
  float* x1       = (float*)alloc((size_t)ROWS * DIM * 4);
  float* offb     = (float*)alloc((size_t)ROWS * 64 * 4);
  float* awlb     = (float*)alloc((size_t)ROWS * 32 * 4);
  float* da_samp  = (float*)alloc((size_t)ROWS * DIM * 4);
  float* da_proj  = (float*)alloc((size_t)ROWS * DIM * 4);
  float* x2       = (float*)alloc((size_t)ROWS * DIM * 4);
  float* ff1      = (float*)alloc((size_t)ROWS * DFF * 4);
  float* ff2      = (float*)alloc((size_t)ROWS * DIM * 4);
  unsigned short* value_bf = (unsigned short*)alloc((size_t)NBATCH * HW * DIM * 2);

  const dim3 blk256(256), blk128(128), blk32(32);

  // --- self attention ---
  gemm_wide_kernel<false, false, false><<<dim3(ROWS / 32, DIM / 128), blk256, 0, stream>>>(
      queries, sa_wq, sa_bq, qp, ROWS, DIM, DIM);
  gemm_wide_kernel<false, false, false><<<dim3(ROWS / 32, DIM / 128), blk256, 0, stream>>>(
      queries, sa_wk, sa_bk, kp, ROWS, DIM, DIM);
  gemm_wide_kernel<false, false, false><<<dim3(ROWS / 32, DIM / 128), blk256, 0, stream>>>(
      queries, sa_wv, sa_bv, vp, ROWS, DIM, DIM);
  attn_kernel<<<dim3(KTILES, NBATCH * NHEAD), blk32, 0, stream>>>(qp, kp, vp, attn_out);
  gemm_wide_kernel<false, false, false><<<dim3(ROWS / 32, DIM / 128), blk256, 0, stream>>>(
      attn_out, sa_wo, sa_bo, sa_proj, ROWS, DIM, DIM);
  resid_ln_kernel<<<dim3(ROWS / 8), blk256, 0, stream>>>(queries, sa_proj, ln1_g, ln1_b, x1, ROWS);

  // --- deformable attention ---
  // bev is streamed once (327 MB): non-temporal loads keep L2 (192 MB) free
  // for the bf16 value tensor (164 MB) that the gather re-reads randomly.
  gemm_wide_kernel<false, true, true><<<dim3(NBATCH * HW / 32, DIM / 128), blk256, 0, stream>>>(
      bev, da_vw, da_vb, value_bf, NBATCH * HW, DIM, DIM);
  gemm_small_kernel<<<dim3(ROWS / 32, 64 / 32), blk128, 0, stream>>>(
      x1, da_ow, da_ob, offb, ROWS, 64, DIM);
  gemm_small_kernel<<<dim3(ROWS / 32, 32 / 32), blk128, 0, stream>>>(
      x1, da_aw, da_ab, awlb, ROWS, 32, DIM);
  deform_sample_kernel<<<dim3(ROWS * NHEAD / 4), blk128, 0, stream>>>(
      ref, offb, awlb, value_bf, da_samp);
  gemm_wide_kernel<false, false, false><<<dim3(ROWS / 32, DIM / 128), blk256, 0, stream>>>(
      da_samp, da_out_w, da_out_b, da_proj, ROWS, DIM, DIM);
  resid_ln_kernel<<<dim3(ROWS / 8), blk256, 0, stream>>>(x1, da_proj, ln2_g, ln2_b, x2, ROWS);

  // --- feedforward ---
  gemm_wide_kernel<true, false, false><<<dim3(ROWS / 32, DFF / 128), blk256, 0, stream>>>(
      x2, ff_w1, ff_b1, ff1, ROWS, DFF, DIM);
  gemm_wide_kernel<false, false, false><<<dim3(ROWS / 32, DIM / 128), blk256, 0, stream>>>(
      ff1, ff_w2, ff_b2, ff2, ROWS, DIM, DFF);
  resid_ln_kernel<<<dim3(ROWS / 8), blk256, 0, stream>>>(x2, ff2, ln3_g, ln3_b,
                                                         (float*)d_out, ROWS);
}
